// StickBreakingVAE_32289564131394
// MI455X (gfx1250) — compile-verified, matrix operand order checked by reading, not run on hardware
//
#include <hip/hip_runtime.h>
#include <hip/hip_bf16.h>
#include <hip/hip_fp16.h>

// ---------------------------------------------------------------------------
// StickBreakingVAE forward, MI455X (gfx1250, wave32, WMMA).
// Big GEMMs (x@W1^T, h2@W3^T) and the pi@W2^T GEMM run on
// v_wmma_f32_16x16x32_f16 with 32x64 strips per wave (8 wmma chains / k-step)
// and global_prefetch_b8 ahead of the streamed weight rows.
// Encoder latent head + Kumaraswamy sample + stick-breaking cumprod run in a
// small per-row LDS kernel.
// ---------------------------------------------------------------------------

typedef __attribute__((ext_vector_type(16))) _Float16 v16h;
typedef __attribute__((ext_vector_type(8)))  _Float16 v8h;
typedef __attribute__((ext_vector_type(4)))  _Float16 v4h;
typedef __attribute__((ext_vector_type(8)))  float    v8f;

#define BATCH       16384
#define INPUT_NDIMS 1024
#define HIDDEN      512
#define LATENT      50
#define LATENT_PAD  64

union Frag {
    v16h v;
    v8h  h[2];
};

__device__ __forceinline__ v8h load8h(const _Float16* p) {
    return *reinterpret_cast<const v8h*>(p);
}

// ---------------------------------------------------------------------------
// vectorized f32 -> f16 conversion (n must be a multiple of 4; all our sizes are)
// ---------------------------------------------------------------------------
__global__ void cvt_f32_f16_v4(const float* __restrict__ in,
                               _Float16* __restrict__ out, int n4) {
    int i = blockIdx.x * blockDim.x + threadIdx.x;
    if (i < n4) {
        float4 f = reinterpret_cast<const float4*>(in)[i];
        v4h o = {(_Float16)f.x, (_Float16)f.y, (_Float16)f.z, (_Float16)f.w};
        reinterpret_cast<v4h*>(out)[i] = o;
    }
}

// W2 [HIDDEN, LATENT] f32 -> [HIDDEN, LATENT_PAD] f16, zero padded in k
__global__ void cvt_pad_w2(const float* __restrict__ w2,
                           _Float16* __restrict__ out) {
    int i = blockIdx.x * blockDim.x + threadIdx.x;
    if (i < HIDDEN * LATENT_PAD) {
        int row = i / LATENT_PAD;
        int k   = i % LATENT_PAD;
        out[i] = (k < LATENT) ? (_Float16)w2[row * LATENT + k] : (_Float16)0.f;
    }
}

// ---------------------------------------------------------------------------
// WMMA GEMM:  C[m,n] = act( sum_k A[m,k]*B[n,k] + bias[n] )
//   A: [M,K] f16 row-major (k contiguous)   B: [N,K] f16 row-major
//   Block = 256 threads = 8 waves. Each block: 32 rows x 512 cols.
//   Each wave: 32(M) x 64(N) strip = 8 WMMA accumulators, 8 wmma / k-step.
//   grid = (M/32, N/512)
// A/B fragment layout per CDNA5 ISA (16-bit A 16x32): lane L holds row L&15;
// two contiguous 8xf16 runs at k0+(L>>4)*8 and k0+16+(L>>4)*8 -> two b128 loads.
// C/D layout: lane L owns column n = L&15 (+tile), VGPR j -> row j + 8*(L>>4).
// ACT: 0 = relu, f16 out ; 1 = sigmoid, f32 out
// ---------------------------------------------------------------------------
template <int ACT>
__global__ void __launch_bounds__(256)
wmma_gemm(const _Float16* __restrict__ A,
          const _Float16* __restrict__ Bw,
          const float* __restrict__ bias,
          void* __restrict__ Out,
          int M, int N, int K) {
    const int lane = threadIdx.x & 31;
    const int wave = threadIdx.x >> 5;
    const int g    = lane >> 4;    // lane half (K sub-block select)
    const int r    = lane & 15;    // row (A) / col (B) within tile

    const int m0     = blockIdx.x * 32;
    const int n_base = blockIdx.y * 512 + wave * 64;
    if (m0 >= M || n_base >= N) return;   // wave-uniform guard

    const _Float16* Arow0 = A  + (size_t)(m0 + r)      * K;
    const _Float16* Arow1 = A  + (size_t)(m0 + 16 + r) * K;
    const _Float16* Brow0 = Bw + (size_t)(n_base + r)  * K;

    v8f acc[2][4] = {};

    for (int k0 = 0; k0 < K; k0 += 32) {
        Frag a0, a1;
        a0.h[0] = load8h(Arow0 + k0 +      g * 8);
        a0.h[1] = load8h(Arow0 + k0 + 16 + g * 8);
        a1.h[0] = load8h(Arow1 + k0 +      g * 8);
        a1.h[1] = load8h(Arow1 + k0 + 16 + g * 8);
#pragma unroll
        for (int t = 0; t < 4; ++t) {
            const _Float16* Brow = Brow0 + (size_t)(t * 16) * K;
            // speculative prefetch of this row's next cachelines
            // (lowers to global_prefetch_b8; OOB at the K tail is dropped)
            __builtin_prefetch(Brow + k0 + 128, 0, 1);
            Frag b;
            b.h[0] = load8h(Brow + k0 +      g * 8);
            b.h[1] = load8h(Brow + k0 + 16 + g * 8);
            // 8 args: (neg_a, A, neg_b, B, c_mod, C, reuse_a, reuse_b)
            acc[0][t] = __builtin_amdgcn_wmma_f32_16x16x32_f16(
                false, a0.v, false, b.v, (short)0, acc[0][t], false, false);
            acc[1][t] = __builtin_amdgcn_wmma_f32_16x16x32_f16(
                false, a1.v, false, b.v, (short)0, acc[1][t], false, false);
        }
    }

#pragma unroll
    for (int i = 0; i < 2; ++i) {
#pragma unroll
        for (int t = 0; t < 4; ++t) {
            const int   n  = n_base + t * 16 + r;  // fixed column per lane
            const float bv = bias[n];
#pragma unroll
            for (int j = 0; j < 8; ++j) {
                const int m   = m0 + i * 16 + j + 8 * g;
                float     val = acc[i][t][j] + bv;
                if (ACT == 0) {
                    val = fmaxf(val, 0.f);
                    ((_Float16*)Out)[(size_t)m * N + n] = (_Float16)val;
                } else {
                    val = 1.f / (1.f + __expf(-val));
                    ((float*)Out)[(size_t)m * N + n] = val;
                }
            }
        }
    }
}

// ---------------------------------------------------------------------------
// Per-row latent head: alpha/beta = softplus(h@Wa^T/Wb^T + b) + 1e-4,
// Kumaraswamy sample v = (1 - u^(1/beta))^(1/alpha), v[K-1]=1,
// stick-breaking pi[k] = v[k] * prod_{j<k}(1 - v[j]); pi stored f16, padded.
// One block (128 threads) per batch row.
// ---------------------------------------------------------------------------
__global__ void __launch_bounds__(128)
encode_latent(const _Float16* __restrict__ h,   // [B, HIDDEN] f16
              const float* __restrict__ u,      // [B, LATENT]
              const float* __restrict__ Wa, const float* __restrict__ ba,
              const float* __restrict__ Wb, const float* __restrict__ bb,
              float* __restrict__ alpha_out,    // [B, LATENT]
              float* __restrict__ beta_out,     // [B, LATENT]
              _Float16* __restrict__ pi_out) {  // [B, LATENT_PAD]
    __shared__ float sh[HIDDEN];
    __shared__ float sab[2 * LATENT];
    __shared__ float sv[LATENT];

    const int row = blockIdx.x;
    const int t   = threadIdx.x;

    for (int i = t; i < HIDDEN; i += 128)
        sh[i] = (float)h[(size_t)row * HIDDEN + i];
    __syncthreads();

    if (t < 2 * LATENT) {
        const int    which = t / LATENT;  // 0 = alpha, 1 = beta
        const int    k     = t % LATENT;
        const float* wr    = (which ? Wb : Wa) + (size_t)k * HIDDEN;
        float        acc   = which ? bb[k] : ba[k];
        for (int i = 0; i < HIDDEN; ++i) acc = fmaf(sh[i], wr[i], acc);
        float sp = (acc > 20.f) ? acc : log1pf(__expf(acc));
        sp += 1e-4f;
        sab[t] = sp;
        if (which) beta_out[(size_t)row * LATENT + k]  = sp;
        else       alpha_out[(size_t)row * LATENT + k] = sp;
    }
    __syncthreads();

    if (t < LATENT) {
        const float a  = sab[t];
        const float b  = sab[LATENT + t];
        const float uu = u[(size_t)row * LATENT + t];
        float v = powf(1.f - powf(uu, 1.f / b), 1.f / a);
        if (t == LATENT - 1) v = 1.f;
        sv[t] = v;
    }
    __syncthreads();

    if (t == 0) {
        float cp = 1.f;
        for (int k = 0; k < LATENT; ++k) {
            const float pi = sv[k] * cp;
            cp *= (1.f - sv[k]);
            pi_out[(size_t)row * LATENT_PAD + k] = (_Float16)pi;
        }
        for (int k = LATENT; k < LATENT_PAD; ++k)
            pi_out[(size_t)row * LATENT_PAD + k] = (_Float16)0.f;
    }
}

// ---------------------------------------------------------------------------
extern "C" void kernel_launch(void* const* d_in, const int* in_sizes, int n_in,
                              void* d_out, int out_size, void* d_ws,
                              size_t ws_size, hipStream_t stream) {
    (void)in_sizes; (void)n_in; (void)out_size; (void)ws_size;

    const float* x  = (const float*)d_in[0];
    const float* u  = (const float*)d_in[1];
    const float* W1 = (const float*)d_in[2];
    const float* b1 = (const float*)d_in[3];
    const float* Wa = (const float*)d_in[4];
    const float* ba = (const float*)d_in[5];
    const float* Wb = (const float*)d_in[6];
    const float* bb = (const float*)d_in[7];
    const float* W2 = (const float*)d_in[8];
    const float* b2 = (const float*)d_in[9];
    const float* W3 = (const float*)d_in[10];
    const float* b3 = (const float*)d_in[11];

    float* recon = (float*)d_out;                              // [B, D]
    float* alpha = recon + (size_t)BATCH * INPUT_NDIMS;        // [B, K]
    float* beta  = alpha + (size_t)BATCH * LATENT;             // [B, K]

    char* ws = (char*)d_ws;
    _Float16* xh   = (_Float16*)ws; ws += (size_t)BATCH * INPUT_NDIMS * 2;
    _Float16* W1h  = (_Float16*)ws; ws += (size_t)HIDDEN * INPUT_NDIMS * 2;
    _Float16* hbuf = (_Float16*)ws; ws += (size_t)BATCH * HIDDEN * 2;
    _Float16* pibf = (_Float16*)ws; ws += (size_t)BATCH * LATENT_PAD * 2;
    _Float16* W2h  = (_Float16*)ws; ws += (size_t)HIDDEN * LATENT_PAD * 2;
    _Float16* h2bf = (_Float16*)ws; ws += (size_t)BATCH * HIDDEN * 2;
    _Float16* W3h  = (_Float16*)ws; ws += (size_t)INPUT_NDIMS * HIDDEN * 2;

    // --- precision conversions (f32 -> f16), 4-wide ---
    {
        int n4 = (BATCH * INPUT_NDIMS) / 4;
        cvt_f32_f16_v4<<<(n4 + 255) / 256, 256, 0, stream>>>(x, xh, n4);
    }
    {
        int n4 = (HIDDEN * INPUT_NDIMS) / 4;
        cvt_f32_f16_v4<<<(n4 + 255) / 256, 256, 0, stream>>>(W1, W1h, n4);
    }
    {
        int n4 = (INPUT_NDIMS * HIDDEN) / 4;
        cvt_f32_f16_v4<<<(n4 + 255) / 256, 256, 0, stream>>>(W3, W3h, n4);
    }
    {
        int n = HIDDEN * LATENT_PAD;
        cvt_pad_w2<<<(n + 255) / 256, 256, 0, stream>>>(W2, W2h);
    }

    // --- GEMM1: h = relu(x @ W1^T + b1)  [16384, 512], K=1024 ---
    wmma_gemm<0><<<dim3(BATCH / 32, HIDDEN / 512), 256, 0, stream>>>(
        xh, W1h, b1, (void*)hbuf, BATCH, HIDDEN, INPUT_NDIMS);

    // --- latent head: alpha, beta, v, pi ---
    encode_latent<<<BATCH, 128, 0, stream>>>(hbuf, u, Wa, ba, Wb, bb,
                                             alpha, beta, pibf);

    // --- GEMM3: h2 = relu(pi @ W2^T + b2)  [16384, 512], K=64 (padded) ---
    wmma_gemm<0><<<dim3(BATCH / 32, HIDDEN / 512), 256, 0, stream>>>(
        pibf, W2h, b2, (void*)h2bf, BATCH, HIDDEN, LATENT_PAD);

    // --- GEMM4: recon = sigmoid(h2 @ W3^T + b3)  [16384, 1024], K=512 ---
    wmma_gemm<1><<<dim3(BATCH / 32, INPUT_NDIMS / 512), 256, 0, stream>>>(
        h2bf, W3h, b3, (void*)recon, BATCH, INPUT_NDIMS, HIDDEN);
}
